// PetrovGalerkinHead_67259187855529
// MI455X (gfx1250) — matrix operand order repaired and checked by reading, not compiled, not run on hardware
//
#include <hip/hip_runtime.h>
#include <hip/hip_bf16.h>
#include <cstdint>

// ---------------------------------------------------------------------------
// MI455X / gfx1250: wave32, WMMA 16x16x32 bf16 (D=f32), 320KB LDS per WGP.
// Compute-bound op (~274 GFLOP vs ~150MB traffic @ 23.3TB/s) -> bf16 WMMA.
// ---------------------------------------------------------------------------

typedef __attribute__((ext_vector_type(16))) __bf16 v16bf;
typedef __attribute__((ext_vector_type(8)))  __bf16 v8bf;
typedef __attribute__((ext_vector_type(8)))  float  v8f;

#define DEV __device__ __forceinline__

DEV v8f wmma_bf16(v16bf a, v16bf b, v8f c) {
  return __builtin_amdgcn_wmma_f32_16x16x32_bf16(false, a, false, b, (short)0, c,
                                                 false, false);
}

// A-matrix (16x32 bf16) per-lane fragment from an LDS row (row-major, stride in
// halfs). Lane l: M = l&15 ; halves 0..15 -> K = (l>>4)*16 + 0..15.
// Two ds_load_b128 per lane.
DEV v16bf load_a(const __bf16* rowBase, int kOff, int lane) {
  const int g = (lane >> 4) * 16;
  v8bf lo = *(const v8bf*)(rowBase + kOff + g);
  v8bf hi = *(const v8bf*)(rowBase + kOff + g + 8);
  return __builtin_shufflevector(lo, hi, 0, 1, 2, 3, 4, 5, 6, 7, 8, 9, 10, 11,
                                 12, 13, 14, 15);
}

DEV float gelu_f(float x) {
  float x3 = x * x * x;
  return 0.5f * x * (1.0f + tanhf(0.7978845608028654f * (x + 0.044715f * x3)));
}

DEV float rmax16(float v) {
#pragma unroll
  for (int off = 1; off < 16; off <<= 1) v = fmaxf(v, __shfl_xor(v, off, 32));
  return v;
}
DEV float rsum16(float v) {
#pragma unroll
  for (int off = 1; off < 16; off <<= 1) v += __shfl_xor(v, off, 32);
  return v;
}

// ---------------------------------------------------------------------------
// Weight repack: W[fin][fout] fp32 -> bf16 tiles matching B-matrix (32x16)
// layout: packed[((cb*KB + kb)*32 + lane)*16 + t] = W[kb*32+(lane>>4)*16+t][cb*16+(lane&15)]
// ---------------------------------------------------------------------------
__global__ void pack_w_kernel(const float* __restrict__ src,
                              __bf16* __restrict__ dst, int fin, int fout) {
  int tid = blockIdx.x * blockDim.x + threadIdx.x;
  int KB = fin >> 5;
  int total = (fout >> 4) * KB * 32;
  if (tid >= total) return;
  int lane = tid & 31;
  int rest = tid >> 5;
  int kb = rest % KB;
  int cb = rest / KB;
  int kBase = kb * 32 + (lane >> 4) * 16;
  int col = cb * 16 + (lane & 15);
#pragma unroll
  for (int t = 0; t < 16; ++t)
    dst[(size_t)tid * 16 + t] = (__bf16)src[(size_t)(kBase + t) * fout + col];
}

__global__ void logw_kernel(const float* __restrict__ sw,
                            float* __restrict__ lw, int n) {
  int i = blockIdx.x * blockDim.x + threadIdx.x;
  if (i < n) lw[i] = logf(fmaxf(sw[i], 1e-8f));
}

__global__ void qinit_kernel(const float* __restrict__ qt,
                             float* __restrict__ Qf, __bf16* __restrict__ Qbf,
                             int n) {
  int i = blockIdx.x * blockDim.x + threadIdx.x;
  if (i < n) {
    float v = qt[i & (256 * 128 - 1)];
    Qf[i] = v;
    Qbf[i] = (__bf16)v;
  }
}

// ---------------------------------------------------------------------------
// Generic fused GEMM layer: 128 rows (one per-WG token tile), A bf16 in LDS,
// W packed bf16 in global (L2-hot), bias fp32.  8 waves x 16 rows, column loop
// 4 tiles at a time, K loop of v_wmma_f32_16x16x32_bf16.
// ---------------------------------------------------------------------------
enum { OUT_LDS_BF16_GELU = 0, OUT_LDS_BF16 = 1, OUT_LDS_F32 = 2, OUT_GLOB_F32 = 3 };

template <int FIN, int FOUT, int MODE>
DEV void mlp_layer(const __bf16* __restrict__ aLds, int aStride,
                   const __bf16* __restrict__ wP, const float* __restrict__ bias,
                   __bf16* __restrict__ outBf, float* __restrict__ outF) {
  const int tid = threadIdx.x;
  const int lane = tid & 31;
  const int w = tid >> 5;
  const int laneCol = lane & 15;
  const int halfSel = lane >> 4;
  const __bf16* aRow = aLds + (size_t)(w * 16 + laneCol) * aStride;
  constexpr int KB = FIN / 32;
  constexpr int CT = FOUT / 16;
  for (int cb = 0; cb < CT; cb += 4) {
    v8f acc[4] = {};
    for (int kb = 0; kb < KB; ++kb) {
      v16bf a = load_a(aRow, kb * 32, lane);
#pragma unroll
      for (int j = 0; j < 4; ++j) {
        const v16bf bt = *(const v16bf*)(wP + (((size_t)(cb + j) * KB + kb) * 32 + lane) * 16);
        acc[j] = wmma_bf16(a, bt, acc[j]);
      }
    }
#pragma unroll
    for (int j = 0; j < 4; ++j) {
      int col = (cb + j) * 16 + laneCol;
      float bv = bias[col];
#pragma unroll
      for (int r = 0; r < 8; ++r) {
        int row = w * 16 + halfSel * 8 + r;  // C layout: lanes>=16 hold M=8..15
        float v = acc[j][r] + bv;
        if constexpr (MODE == OUT_LDS_BF16_GELU) v = gelu_f(v);
        if constexpr (MODE == OUT_LDS_BF16_GELU || MODE == OUT_LDS_BF16)
          outBf[(size_t)row * FOUT + col] = (__bf16)v;
        else
          outF[(size_t)row * FOUT + col] = v;  // LDS f32 or global f32
      }
    }
  }
}

// ---------------------------------------------------------------------------
// K/V MLP3 kernel: 128 tokens per WG.  LDS: xu(128x192) + h1(128x512) +
// h2(128x512) bf16 = 304KB.  All intermediates stay in LDS.
// ---------------------------------------------------------------------------
__global__ __launch_bounds__(256) void kv_kernel(
    const float* __restrict__ x, const float* __restrict__ u,
    const __bf16* pkK1, const float* bK1, const __bf16* pkK2, const float* bK2,
    const __bf16* pkK3, const float* bK3, const __bf16* pkV1, const float* bV1,
    const __bf16* pkV2, const float* bV2, const __bf16* pkV3, const float* bV3,
    __bf16* __restrict__ Kbf, __bf16* __restrict__ Vt) {
  extern __shared__ char smem[];
  __bf16* xu = (__bf16*)smem;              // 128 x 192
  __bf16* h1 = xu + 128 * 192;             // 128 x 512
  __bf16* h2 = h1 + 128 * 512;             // 128 x 512
  const int tid = threadIdx.x;
  const size_t tok0 = (size_t)blockIdx.x * 128;
  const int bidx = (int)(tok0 >> 12);      // N = 4096
  const int nbase = (int)(tok0 & 4095);

  // Stage x_enc||u tile -> LDS bf16
  for (int i = tid; i < 128 * 192; i += 256) {
    int row = i / 192, c = i % 192;
    float v = (c < 128) ? x[(tok0 + row) * 128 + c] : u[(tok0 + row) * 64 + (c - 128)];
    xu[(size_t)row * 192 + c] = (__bf16)v;
  }
  __syncthreads();

  // ---- K = mlp3(x) ----
  mlp_layer<128, 512, OUT_LDS_BF16_GELU>(xu, 192, pkK1, bK1, h1, nullptr);
  __syncthreads();
  mlp_layer<512, 512, OUT_LDS_BF16_GELU>(h1, 512, pkK2, bK2, h2, nullptr);
  __syncthreads();
  mlp_layer<512, 128, OUT_LDS_BF16>(h2, 512, pkK3, bK3, h1, nullptr);
  __syncthreads();
  {  // copy K tile (128x128) to global, row-major
    v8bf* dst = (v8bf*)(Kbf + tok0 * 128);
    const v8bf* src = (const v8bf*)h1;
    for (int i = tid; i < 128 * 16; i += 256) dst[i] = src[i];
  }
  __syncthreads();

  // ---- V = mlp3([x,u]) ----
  mlp_layer<192, 512, OUT_LDS_BF16_GELU>(xu, 192, pkV1, bV1, h1, nullptr);
  __syncthreads();
  mlp_layer<512, 512, OUT_LDS_BF16_GELU>(h1, 512, pkV2, bV2, h2, nullptr);
  __syncthreads();
  mlp_layer<512, 128, OUT_LDS_BF16>(h2, 512, pkV3, bV3, h1, nullptr);
  __syncthreads();
  {  // transposed store: Vt[b*128+dv][n]
    int dv = tid & 127, hh = tid >> 7;
    for (int rr = 0; rr < 64; rr += 8) {
      v8bf v;
#pragma unroll
      for (int j = 0; j < 8; ++j) v[j] = h1[(size_t)(hh * 64 + rr + j) * 128 + dv];
      *(v8bf*)(Vt + ((size_t)bidx * 128 + dv) * 4096 + nbase + hh * 64 + rr) = v;
    }
  }
}

// ---------------------------------------------------------------------------
// Flash-style attention: scores = Q K^T * scale + log_w ; softmax over N ;
// pooled = A V.  Each wave: 16 query rows, full DV=128, N streamed in 32-chunks
// with online softmax.  16 WMMAs / chunk / wave.
// ---------------------------------------------------------------------------
__global__ __launch_bounds__(256) void attn_kernel(
    const __bf16* __restrict__ Qbf, const __bf16* __restrict__ Kbf,
    const __bf16* __restrict__ Vt, const float* __restrict__ logw,
    const unsigned char* __restrict__ mask, __bf16* __restrict__ pooledbf) {
  extern __shared__ char smem[];
  __bf16* qLds = (__bf16*)smem;                  // 128 x 128 bf16 (32KB)
  __bf16* pstage = qLds + 128 * 128;             // 8 waves x 16 x 32 (8KB)
  const int tid = threadIdx.x;
  const int lane = tid & 31, w = tid >> 5;
  const int laneCol = lane & 15, halfSel = lane >> 4;
  const int b = blockIdx.x >> 1;
  const int pblk = blockIdx.x & 1;
  const float scale = 0.08838834764831845f;  // 1/sqrt(128)

  {  // stage Q tile
    const v8bf* src = (const v8bf*)(Qbf + ((size_t)b * 256 + pblk * 128) * 128);
    v8bf* dst = (v8bf*)qLds;
    for (int i = tid; i < 128 * 16; i += 256) dst[i] = src[i];
  }
  __syncthreads();

  const __bf16* aRow = qLds + (size_t)(w * 16 + laneCol) * 128;
  __bf16* st = pstage + (size_t)w * 16 * 32;

  v8f acc[8] = {};
  float m_run[8], l_run[8];
#pragma unroll
  for (int r = 0; r < 8; ++r) { m_run[r] = -1e30f; l_run[r] = 0.f; }

  for (int nc = 0; nc < 128; ++nc) {
    const int n0 = nc * 32;
    const __bf16* kb0 = Kbf + ((size_t)b * 4096 + n0 + laneCol) * 128 + halfSel * 16;
    const __bf16* kb1 = kb0 + (size_t)16 * 128;
    if (nc + 1 < 128) __builtin_prefetch(kb0 + (size_t)32 * 128, 0, 0);

    v8f S0 = {}, S1 = {};
#pragma unroll
    for (int kb = 0; kb < 4; ++kb) {
      v16bf a = load_a(aRow, kb * 32, lane);
      v16bf b0 = *(const v16bf*)(kb0 + kb * 32);
      v16bf b1 = *(const v16bf*)(kb1 + kb * 32);
      S0 = wmma_bf16(a, b0, S0);
      S1 = wmma_bf16(a, b1, S1);
    }
    const float lw0 = logw[(size_t)b * 4096 + n0 + laneCol];
    const float lw1 = logw[(size_t)b * 4096 + n0 + 16 + laneCol];
    const bool mk0 = mask[(size_t)b * 4096 + n0 + laneCol] != 0;
    const bool mk1 = mask[(size_t)b * 4096 + n0 + 16 + laneCol] != 0;

    float s0[8], s1[8], mloc[8];
#pragma unroll
    for (int r = 0; r < 8; ++r) {
      s0[r] = mk0 ? S0[r] * scale + lw0 : -1e30f;
      s1[r] = mk1 ? S1[r] * scale + lw1 : -1e30f;
      mloc[r] = fmaxf(s0[r], s1[r]);
    }
#pragma unroll
    for (int r = 0; r < 8; ++r) mloc[r] = rmax16(mloc[r]);

    float p0[8], p1[8], ef[8];
#pragma unroll
    for (int r = 0; r < 8; ++r) {
      float mnew = fmaxf(m_run[r], mloc[r]);
      ef[r] = expf(m_run[r] - mnew);
      p0[r] = expf(s0[r] - mnew);
      p1[r] = expf(s1[r] - mnew);
      m_run[r] = mnew;
    }
#pragma unroll
    for (int r = 0; r < 8; ++r)
      l_run[r] = l_run[r] * ef[r] + rsum16(p0[r] + p1[r]);
#pragma unroll
    for (int t = 0; t < 8; ++t)
#pragma unroll
      for (int r = 0; r < 8; ++r) acc[t][r] *= ef[r];

    // P (C layout) -> LDS -> A layout (16x32)
    const int rb = halfSel * 8;
#pragma unroll
    for (int r = 0; r < 8; ++r) {
      st[(size_t)(rb + r) * 32 + laneCol] = (__bf16)p0[r];
      st[(size_t)(rb + r) * 32 + 16 + laneCol] = (__bf16)p1[r];
    }
    asm volatile("" ::: "memory");  // keep LDS store->load ordered (in-order DS)
    v16bf pa = load_a(st + (size_t)laneCol * 32, 0, lane);

    const __bf16* vb = Vt + ((size_t)b * 128 + laneCol) * 4096 + n0 + halfSel * 16;
#pragma unroll
    for (int t = 0; t < 8; ++t) {
      v16bf bt = *(const v16bf*)(vb + (size_t)t * 16 * 4096);
      acc[t] = wmma_bf16(pa, bt, acc[t]);
    }
  }

  // normalize + store pooled (bf16)
#pragma unroll
  for (int t = 0; t < 8; ++t) {
    int col = t * 16 + laneCol;
#pragma unroll
    for (int r = 0; r < 8; ++r) {
      int prow = pblk * 128 + w * 16 + halfSel * 8 + r;
      float v = acc[t][r] / l_run[r];
      pooledbf[((size_t)b * 256 + prow) * 128 + col] = (__bf16)v;
    }
  }
}

// ---------------------------------------------------------------------------
// Refinement: delta_Q = mlp2(pooled) ; Q = LayerNorm(Q + delta_Q)
// ---------------------------------------------------------------------------
__global__ __launch_bounds__(256) void refine_kernel(
    const __bf16* __restrict__ pooledbf, const __bf16* pkR1, const float* bR1,
    const __bf16* pkR2, const float* bR2, const float* __restrict__ lnG,
    const float* __restrict__ lnB, float* __restrict__ Qf,
    __bf16* __restrict__ Qbf) {
  extern __shared__ char smem[];
  __bf16* aBf = (__bf16*)smem;        // 128x128 bf16
  __bf16* h1 = aBf + 128 * 128;       // 128x512 bf16
  float* dbuf = (float*)(h1 + 128 * 512);  // 128x128 f32
  const int tid = threadIdx.x;
  const size_t rowbase = (size_t)blockIdx.x * 128;

  {
    const v8bf* src = (const v8bf*)(pooledbf + rowbase * 128);
    v8bf* dst = (v8bf*)aBf;
    for (int i = tid; i < 128 * 16; i += 256) dst[i] = src[i];
  }
  __syncthreads();
  mlp_layer<128, 512, OUT_LDS_BF16_GELU>(aBf, 128, pkR1, bR1, h1, nullptr);
  __syncthreads();
  mlp_layer<512, 128, OUT_LDS_F32>(h1, 512, pkR2, bR2, nullptr, dbuf);
  __syncthreads();

  if (tid < 128) {
    const size_t grow = rowbase + tid;
    float s = 0.f, s2 = 0.f;
    for (int c = 0; c < 128; ++c) {
      float v = Qf[grow * 128 + c] + dbuf[(size_t)tid * 128 + c];
      s += v;
      s2 += v * v;
    }
    float mean = s * (1.0f / 128.0f);
    float var = s2 * (1.0f / 128.0f) - mean * mean;
    float rs = rsqrtf(var + 1e-5f);
    for (int c = 0; c < 128; ++c) {
      float v = Qf[grow * 128 + c] + dbuf[(size_t)tid * 128 + c];
      float o = (v - mean) * rs * lnG[c] + lnB[c];
      Qf[grow * 128 + c] = o;
      Qbf[grow * 128 + c] = (__bf16)o;
    }
  }
}

// ---------------------------------------------------------------------------
// Output head: out = mlp2(pooled) -> fp32 d_out (B,P,256)
// ---------------------------------------------------------------------------
__global__ __launch_bounds__(256) void rho_kernel(
    const __bf16* __restrict__ pooledbf, const __bf16* pkW1, const float* bW1,
    const __bf16* pkW2, const float* bW2, float* __restrict__ out) {
  extern __shared__ char smem[];
  __bf16* aBf = (__bf16*)smem;   // 128x128
  __bf16* h1 = aBf + 128 * 128;  // 128x512
  const int tid = threadIdx.x;
  const size_t rowbase = (size_t)blockIdx.x * 128;
  {
    const v8bf* src = (const v8bf*)(pooledbf + rowbase * 128);
    v8bf* dst = (v8bf*)aBf;
    for (int i = tid; i < 128 * 16; i += 256) dst[i] = src[i];
  }
  __syncthreads();
  mlp_layer<128, 512, OUT_LDS_BF16_GELU>(aBf, 128, pkW1, bW1, h1, nullptr);
  __syncthreads();
  mlp_layer<512, 256, OUT_GLOB_F32>(h1, 512, pkW2, bW2, nullptr,
                                    out + rowbase * 256);
}

// ---------------------------------------------------------------------------
extern "C" void kernel_launch(void* const* d_in, const int* in_sizes, int n_in,
                              void* d_out, int out_size, void* d_ws,
                              size_t ws_size, hipStream_t stream) {
  (void)in_sizes; (void)n_in; (void)out_size; (void)ws_size;
  const float* x_enc = (const float*)d_in[0];
  const float* u = (const float*)d_in[1];
  const unsigned char* mask = (const unsigned char*)d_in[2];
  const float* sweights = (const float*)d_in[3];
  const float* k_w1 = (const float*)d_in[4];  const float* k_b1 = (const float*)d_in[5];
  const float* k_w2 = (const float*)d_in[6];  const float* k_b2 = (const float*)d_in[7];
  const float* k_w3 = (const float*)d_in[8];  const float* k_b3 = (const float*)d_in[9];
  const float* v_w1 = (const float*)d_in[10]; const float* v_b1 = (const float*)d_in[11];
  const float* v_w2 = (const float*)d_in[12]; const float* v_b2 = (const float*)d_in[13];
  const float* v_w3 = (const float*)d_in[14]; const float* v_b3 = (const float*)d_in[15];
  const float* qtok = (const float*)d_in[16];
  const float* rho_w1 = (const float*)d_in[17]; const float* rho_b1 = (const float*)d_in[18];
  const float* rho_w2 = (const float*)d_in[19]; const float* rho_b2 = (const float*)d_in[20];
  const float* ref_w1 = (const float*)d_in[21]; const float* ref_b1 = (const float*)d_in[22];
  const float* ref_w2 = (const float*)d_in[23]; const float* ref_b2 = (const float*)d_in[24];
  const float* ln_g = (const float*)d_in[25]; const float* ln_b = (const float*)d_in[26];

  const size_t B = 32, N = 4096, P = 256, DK = 128, DV = 128, DX = 128,
               DU = 64, H = 512, DOUT = 256;
  char* ws = (char*)d_ws;
  size_t off = 0;
  auto take = [&](size_t bytes) {
    size_t o = off;
    off += (bytes + 255) & ~(size_t)255;
    return o;
  };
  const size_t oKbf = take(B * N * DK * 2);
  const size_t oVt = take(B * N * DV * 2);
  const size_t oQf = take(B * P * DK * 4);
  const size_t oQbf = take(B * P * DK * 2);
  const size_t oPool = take(B * P * DV * 2);
  const size_t oLogw = take(B * N * 4);
  const size_t oK1 = take(DX * H * 2), oK2 = take(H * H * 2), oK3 = take(H * DK * 2);
  const size_t oV1 = take((DX + DU) * H * 2), oV2 = take(H * H * 2), oV3 = take(H * DV * 2);
  const size_t oR1 = take(DV * H * 2), oR2 = take(H * DK * 2);
  const size_t oP1 = take(DV * H * 2), oP2 = take(H * DOUT * 2);

  auto bfp = [&](size_t o) { return (__bf16*)(ws + o); };

  auto pack = [&](const float* src, size_t dstOff, int fin, int fout) {
    int total = (fout / 16) * (fin / 32) * 32;
    pack_w_kernel<<<(total + 255) / 256, 256, 0, stream>>>(src, bfp(dstOff), fin, fout);
  };
  pack(k_w1, oK1, DX, H);   pack(k_w2, oK2, H, H);   pack(k_w3, oK3, H, DK);
  pack(v_w1, oV1, DX + DU, H); pack(v_w2, oV2, H, H); pack(v_w3, oV3, H, DV);
  pack(ref_w1, oR1, DV, H); pack(ref_w2, oR2, H, DK);
  pack(rho_w1, oP1, DV, H); pack(rho_w2, oP2, H, DOUT);

  logw_kernel<<<(int)((B * N + 255) / 256), 256, 0, stream>>>(
      sweights, (float*)(ws + oLogw), (int)(B * N));
  qinit_kernel<<<(int)((B * P * DK + 255) / 256), 256, 0, stream>>>(
      qtok, (float*)(ws + oQf), bfp(oQbf), (int)(B * P * DK));

  // Fused K/V MLP3: 1024 WGs, 304KB LDS each.
  const size_t kvLds = (128 * 192 + 128 * 512 + 128 * 512) * 2;
  kv_kernel<<<(int)(B * N / 128), 256, kvLds, stream>>>(
      x_enc, u, bfp(oK1), k_b1, bfp(oK2), k_b2, bfp(oK3), k_b3, bfp(oV1), v_b1,
      bfp(oV2), v_b2, bfp(oV3), v_b3, bfp(oKbf), bfp(oVt));

  const size_t attnLds = 128 * 128 * 2 + 8 * 16 * 32 * 2;
  const size_t refLds = 128 * 128 * 2 + 128 * 512 * 2 + 128 * 128 * 4;
  for (int it = 0; it < 3; ++it) {
    attn_kernel<<<(int)(B * P / 128), 256, attnLds, stream>>>(
        bfp(oQbf), bfp(oKbf), bfp(oVt), (float*)(ws + oLogw), mask, bfp(oPool));
    if (it < 2) {
      refine_kernel<<<(int)(B * P / 128), 256, refLds, stream>>>(
          bfp(oPool), bfp(oR1), ref_b1, bfp(oR2), ref_b2, ln_g, ln_b,
          (float*)(ws + oQf), bfp(oQbf));
    }
  }

  const size_t rhoLds = 128 * 128 * 2 + 128 * 512 * 2;
  rho_kernel<<<(int)(B * P / 128), 256, rhoLds, stream>>>(
      bfp(oPool), bfp(oP1), rho_b1, bfp(oP2), rho_b2, (float*)d_out);
}
